// TalkingHeadAttention_11390253269426
// MI455X (gfx1250) — matrix operand order repaired and checked by reading, not compiled, not run on hardware
//
#include <hip/hip_runtime.h>
#include <hip/hip_bf16.h>

#define EMBED 1024
#define HEADS 16
#define HDIM 64
#define SEQ 2048
#define BSZ 2
#define MROWS (SEQ * BSZ)   // 4096

typedef __attribute__((ext_vector_type(16))) __bf16 bf16x16;
typedef __attribute__((ext_vector_type(8)))  float  v8f;
typedef unsigned int u32x4 __attribute__((ext_vector_type(4)));
typedef int i32x4 __attribute__((ext_vector_type(4)));
typedef int i32x8 __attribute__((ext_vector_type(8)));

union ABf16 {
    bf16x16 v;
    uint4   u[2];
    __bf16  h[16];
};

// ---------------------------------------------------------------------------
// Operand loaders.
// A-matrix (16x32 bf16): lane L<16 -> row L, K in {0..7, 16..23};
//                        lane L+16 -> row L, K in {8..15, 24..31}.
// B-matrix (32x16 bf16): lane n<16 -> col n, K in {0..7, 16..23};
//                        lane n+16 -> col n, K in {8..15, 24..31}.
// ---------------------------------------------------------------------------
__device__ __forceinline__ bf16x16 load_a_bf16(const __bf16* rowptr, int k0) {
    const int l = threadIdx.x & 31;
    const int base = k0 + ((l & 16) ? 8 : 0);
    ABf16 r;
    r.u[0] = *(const uint4*)(rowptr + base);
    r.u[1] = *(const uint4*)(rowptr + base + 16);
    return r.v;
}

__device__ __forceinline__ bf16x16 load_b_bf16(const __bf16* rowptr, int k0) {
    const int l = threadIdx.x & 31;
    const int base = k0 + ((l & 16) ? 8 : 0);
    ABf16 r;
    r.u[0] = *(const uint4*)(rowptr + base);
    r.u[1] = *(const uint4*)(rowptr + base + 16);
    return r.v;
}

__device__ __forceinline__ bf16x16 load_a_f32(const float* rowptr, int k0) {
    const int l = threadIdx.x & 31;
    const int base = k0 + ((l & 16) ? 8 : 0);
    ABf16 r;
#pragma unroll
    for (int j = 0; j < 8; ++j) r.h[j] = (__bf16)rowptr[base + j];
#pragma unroll
    for (int j = 0; j < 8; ++j) r.h[8 + j] = (__bf16)rowptr[base + 16 + j];
    return r.v;
}

__device__ __forceinline__ bf16x16 load_b_f32(const float* rowptr, int k0) {
    const int l = threadIdx.x & 31;
    const int base = k0 + ((l & 16) ? 8 : 0);
    ABf16 r;
#pragma unroll
    for (int j = 0; j < 8; ++j) r.h[j] = (__bf16)rowptr[base + j];
#pragma unroll
    for (int j = 0; j < 8; ++j) r.h[8 + j] = (__bf16)rowptr[base + 16 + j];
    return r.v;
}

#define WMMA_BF16(A, B, C) \
    __builtin_amdgcn_wmma_f32_16x16x32_bf16(false, (A), false, (B), (short)0, (C), false, false)

// ---------------------------------------------------------------------------
// Tensor Data Mover: 2-D tile (32 rows x 64 bf16, row stride 64) -> LDS.
// D# packing per CDNA5 ISA.
// ---------------------------------------------------------------------------
__device__ __forceinline__ unsigned lds_off(const void* p) {
    return (unsigned)(unsigned long long)p;  // generic LDS addr: offset in [31:0]
}

__device__ __forceinline__ void tdm_load_k32x64(const __bf16* gtile, unsigned ldsaddr) {
    const unsigned long long ga = (unsigned long long)gtile;
    u32x4 g0;
    g0[0] = 1u;                                    // count=1, user descriptor
    g0[1] = ldsaddr;                               // lds_addr (bytes)
    g0[2] = (unsigned)(ga & 0xffffffffu);          // global_addr[31:0]
    g0[3] = (unsigned)((ga >> 32) & 0x01ffffffu)   // global_addr[56:32]
            | (2u << 30);                          // type = 2 ("image")
    i32x8 g1;
    g1[0] = 0x00010000;                            // data_size=1 (2 bytes/elem)
    g1[1] = (int)(HDIM << 16);                     // tensor_dim0 = 64 (lo16)
    g1[2] = (int)(SEQ << 16);                      // dim0 hi=0 | tensor_dim1 lo16
    g1[3] = (int)(HDIM << 16);                     // dim1 hi=0 | tile_dim0 = 64
    g1[4] = 32;                                    // tile_dim1 = 32, tile_dim2 = 0
    g1[5] = HDIM;                                  // tensor_dim0_stride = 64
    g1[6] = 0;
    g1[7] = 0;
    i32x4 z4 = {0, 0, 0, 0};
#if __clang_major__ >= 23
    i32x8 z8 = {0, 0, 0, 0, 0, 0, 0, 0};
    __builtin_amdgcn_tensor_load_to_lds(g0, g1, z4, z4, z8, 0);
#else
    __builtin_amdgcn_tensor_load_to_lds(g0, g1, z4, z4, 0);
#endif
}

// ---------------------------------------------------------------------------
// Projection: Y = (X @ W^T + bias) * scale, stored bf16.
// mode 0: out[b][h][s][d]   (q / k)
// mode 1: out[b][h][d][s]   (v transposed for the attn B operand)
// Block: 256 threads = 8 waves; wave tile 64x64 (4 cvt-packs per WMMA);
// block tile 128(M) x 256(N).
// ---------------------------------------------------------------------------
__global__ __launch_bounds__(256) void proj_kernel(
    const float* __restrict__ X, const float* __restrict__ W,
    const float* __restrict__ bias, __bf16* __restrict__ out,
    float scale, int mode) {
    const int l = threadIdx.x & 31;
    const int w = threadIdx.x >> 5;
    const int m_base = blockIdx.y * 128 + (w & 1) * 64;
    const int n_base = blockIdx.x * 256 + (w >> 1) * 64;
    const int n = l & 15;
    const int hi = (l >> 4) & 1;

    const float* arow[4];
    const float* brow[4];
#pragma unroll
    for (int i = 0; i < 4; ++i) {
        arow[i] = X + (size_t)(m_base + i * 16 + n) * EMBED;
        brow[i] = W + (size_t)(n_base + i * 16 + n) * EMBED;
    }

    v8f C[4][4] = {};
    for (int k0 = 0; k0 < EMBED; k0 += 32) {
        bf16x16 A[4];
#pragma unroll
        for (int i = 0; i < 4; ++i) A[i] = load_a_f32(arow[i], k0);
#pragma unroll
        for (int j = 0; j < 4; ++j) {
            bf16x16 Bv = load_b_f32(brow[j], k0);
#pragma unroll
            for (int i = 0; i < 4; ++i) C[i][j] = WMMA_BF16(A[i], Bv, C[i][j]);
        }
    }

#pragma unroll
    for (int ai = 0; ai < 4; ++ai)
#pragma unroll
        for (int bi = 0; bi < 4; ++bi)
#pragma unroll
            for (int i = 0; i < 8; ++i) {
                const int row = m_base + ai * 16 + i + 8 * hi;  // m = s*BSZ + b
                const int col = n_base + bi * 16 + n;           // e = h*64 + d
                const float y = (C[ai][bi][i] + bias[col]) * scale;
                const int s = row >> 1, bb = row & 1;
                const int hh = col >> 6, d = col & 63;
                if (mode == 0)
                    out[((size_t)(bb * HEADS + hh) * SEQ + s) * HDIM + d] = (__bf16)y;
                else
                    out[((size_t)(bb * HEADS + hh) * HDIM + d) * SEQ + s] = (__bf16)y;
            }
}

// ---------------------------------------------------------------------------
// Output projection: out = ctx(bf16, row-major MROWSxEMBED) @ Wo^T + bo  (f32)
// ---------------------------------------------------------------------------
__global__ __launch_bounds__(256) void out_proj_kernel(
    const __bf16* __restrict__ X, const float* __restrict__ W,
    const float* __restrict__ bias, float* __restrict__ out) {
    const int l = threadIdx.x & 31;
    const int w = threadIdx.x >> 5;
    const int m_base = blockIdx.y * 128 + (w & 1) * 64;
    const int n_base = blockIdx.x * 256 + (w >> 1) * 64;
    const int n = l & 15;
    const int hi = (l >> 4) & 1;

    const __bf16* arow[4];
    const float* brow[4];
#pragma unroll
    for (int i = 0; i < 4; ++i) {
        arow[i] = X + (size_t)(m_base + i * 16 + n) * EMBED;
        brow[i] = W + (size_t)(n_base + i * 16 + n) * EMBED;
    }

    v8f C[4][4] = {};
    for (int k0 = 0; k0 < EMBED; k0 += 32) {
        bf16x16 A[4];
#pragma unroll
        for (int i = 0; i < 4; ++i) A[i] = load_a_bf16(arow[i], k0);
#pragma unroll
        for (int j = 0; j < 4; ++j) {
            bf16x16 Bv = load_b_f32(brow[j], k0);
#pragma unroll
            for (int i = 0; i < 4; ++i) C[i][j] = WMMA_BF16(A[i], Bv, C[i][j]);
        }
    }

#pragma unroll
    for (int ai = 0; ai < 4; ++ai)
#pragma unroll
        for (int bi = 0; bi < 4; ++bi)
#pragma unroll
            for (int i = 0; i < 8; ++i) {
                const int row = m_base + ai * 16 + i + 8 * hi;
                const int col = n_base + bi * 16 + n;
                out[(size_t)row * EMBED + col] = C[ai][bi][i] + bias[col];
            }
}

// ---------------------------------------------------------------------------
// Fused talking-heads attention. One block = (batch b, 16 query rows) x all
// 16 heads; wave w == head w. K tiles are TDM-prefetched (double-buffered)
// into LDS; scores via WMMA; pre/post head-mixes exchanged through LDS.
// ---------------------------------------------------------------------------
__device__ __forceinline__ void scores_to_lds(
    const __bf16* ktile /* LDS, 32 x HDIM row-major */, bf16x16 qa0, bf16x16 qa1,
    int head, float (*sbuf)[HEADS]) {
    const int l = threadIdx.x & 31;
    const int hi = (l >> 4) & 1;
    const int n = l & 15;
#pragma unroll
    for (int nt = 0; nt < 2; ++nt) {
        const __bf16* krow = ktile + (nt * 16 + n) * HDIM;
        bf16x16 b0 = load_b_bf16(krow, 0);
        bf16x16 b1 = load_b_bf16(krow, 32);
        v8f S = {};
        S = WMMA_BF16(qa0, b0, S);
        S = WMMA_BF16(qa1, b1, S);
#pragma unroll
        for (int i = 0; i < 8; ++i)
            sbuf[nt * 256 + (i + 8 * hi) * 16 + n][head] = S[i];
    }
}

__device__ __forceinline__ void mix_from_lds(
    float (*sbuf)[HEADS], const float* wrow, float* M0, float* M1) {
    const int l = threadIdx.x & 31;
    const int hi = (l >> 4) & 1;
    const int n = l & 15;
#pragma unroll
    for (int i = 0; i < 8; ++i) {
        const int row = i + 8 * hi;
        const int p0 = row * 16 + n;
        const int p1 = 256 + row * 16 + n;
        float a0 = 0.f, a1 = 0.f;
#pragma unroll
        for (int q4 = 0; q4 < 4; ++q4) {
            float4 c0 = *(const float4*)&sbuf[p0][q4 * 4];
            float4 c1 = *(const float4*)&sbuf[p1][q4 * 4];
            a0 += wrow[q4 * 4 + 0] * c0.x + wrow[q4 * 4 + 1] * c0.y +
                  wrow[q4 * 4 + 2] * c0.z + wrow[q4 * 4 + 3] * c0.w;
            a1 += wrow[q4 * 4 + 0] * c1.x + wrow[q4 * 4 + 1] * c1.y +
                  wrow[q4 * 4 + 2] * c1.z + wrow[q4 * 4 + 3] * c1.w;
        }
        M0[i] = a0;
        M1[i] = a1;
    }
}

__device__ __forceinline__ bf16x16 postmix_a(
    float (*sbuf)[HEADS], const float* wpost) {
    const int l = threadIdx.x & 31;
    const int row = l & 15;
    const int cbase = (l & 16) ? 8 : 0;
    ABf16 r;
#pragma unroll
    for (int j = 0; j < 16; ++j) {
        const int c = cbase + (j & 7) + ((j & 8) ? 16 : 0);
        const int pos = (c >> 4) * 256 + row * 16 + (c & 15);
        float acc = 0.f;
#pragma unroll
        for (int q4 = 0; q4 < 4; ++q4) {
            float4 v = *(const float4*)&sbuf[pos][q4 * 4];
            acc += wpost[q4 * 4 + 0] * v.x + wpost[q4 * 4 + 1] * v.y +
                   wpost[q4 * 4 + 2] * v.z + wpost[q4 * 4 + 3] * v.w;
        }
        r.h[j] = (__bf16)acc;
    }
    return r.v;
}

__global__ __launch_bounds__(512) void attn_kernel(
    const __bf16* __restrict__ q_ws, const __bf16* __restrict__ k_ws,
    const __bf16* __restrict__ vT_ws, const float* __restrict__ Wpre,
    const float* __restrict__ Wpost, __bf16* __restrict__ ctx_ws) {
    __shared__ __align__(16) float sbuf[512][HEADS];             // 32 KB
    __shared__ __align__(16) __bf16 kbuf[2][HEADS][32 * HDIM];   // 128 KB

    const int l = threadIdx.x & 31;
    const int head = threadIdx.x >> 5;          // wave id == head id
    const int b = blockIdx.x >> 7;              // SEQ/16 = 128 t-tiles
    const int t0 = (blockIdx.x & 127) << 4;
    const int hi = (l >> 4) & 1;
    const int n = l & 15;

    float wpre[16], wpost[16];
#pragma unroll
    for (int j = 0; j < 16; ++j) {
        wpre[j] = Wpre[head * HEADS + j];
        wpost[j] = Wpost[head * HEADS + j];
    }

    const __bf16* qbase = q_ws + (size_t)(b * HEADS + head) * SEQ * HDIM;
    const __bf16* kbase = k_ws + (size_t)(b * HEADS + head) * SEQ * HDIM;
    const __bf16* vbase = vT_ws + (size_t)(b * HEADS + head) * HDIM * SEQ;

    const __bf16* qrow = qbase + (size_t)(t0 + n) * HDIM;
    bf16x16 qa0 = load_a_bf16(qrow, 0);
    bf16x16 qa1 = load_a_bf16(qrow, 32);

    float mrun[8], lrun[8];
#pragma unroll
    for (int i = 0; i < 8; ++i) { mrun[i] = -3.0e30f; lrun[i] = 0.f; }

    // ---------------- Pass A: softmax statistics ----------------
    tdm_load_k32x64(kbase, lds_off(&kbuf[0][head][0]));
    int cur = 0;
    for (int s0 = 0; s0 < SEQ; s0 += 32, cur ^= 1) {
        if (s0 + 32 < SEQ) {
            tdm_load_k32x64(kbase + (size_t)(s0 + 32) * HDIM,
                            lds_off(&kbuf[cur ^ 1][head][0]));
            __builtin_amdgcn_s_wait_tensorcnt(1);
        } else {
            __builtin_amdgcn_s_wait_tensorcnt(0);
        }
        scores_to_lds(&kbuf[cur][head][0], qa0, qa1, head, sbuf);
        __syncthreads();
        float M0[8], M1[8];
        mix_from_lds(sbuf, wpre, M0, M1);
        __syncthreads();
#pragma unroll
        for (int i = 0; i < 8; ++i) {
            float r = fmaxf(M0[i], M1[i]);
#pragma unroll
            for (int mask = 1; mask <= 8; mask <<= 1)
                r = fmaxf(r, __shfl_xor(r, mask, 32));
            const float mn = fmaxf(mrun[i], r);
            float rs = __expf(M0[i] - mn) + __expf(M1[i] - mn);
#pragma unroll
            for (int mask = 1; mask <= 8; mask <<= 1)
                rs += __shfl_xor(rs, mask, 32);
            lrun[i] = lrun[i] * __expf(mrun[i] - mn) + rs;
            mrun[i] = mn;
        }
    }
    float linv[8];
#pragma unroll
    for (int i = 0; i < 8; ++i) linv[i] = 1.0f / lrun[i];

    // ---------------- Pass B: probabilities + context -----------
    v8f ctx[4] = {};
    tdm_load_k32x64(kbase, lds_off(&kbuf[0][head][0]));
    cur = 0;
    for (int s0 = 0; s0 < SEQ; s0 += 32, cur ^= 1) {
        if (s0 + 32 < SEQ) {
            tdm_load_k32x64(kbase + (size_t)(s0 + 32) * HDIM,
                            lds_off(&kbuf[cur ^ 1][head][0]));
            __builtin_amdgcn_s_wait_tensorcnt(1);
        } else {
            __builtin_amdgcn_s_wait_tensorcnt(0);
        }
        __builtin_prefetch(vbase + (size_t)n * SEQ + s0, 0, 0);
        scores_to_lds(&kbuf[cur][head][0], qa0, qa1, head, sbuf);
        __syncthreads();
        float M0[8], M1[8];
        mix_from_lds(sbuf, wpre, M0, M1);
        __syncthreads();
#pragma unroll
        for (int i = 0; i < 8; ++i) {
            const int row = i + 8 * hi;
            sbuf[row * 16 + n][head] = __expf(M0[i] - mrun[i]) * linv[i];
            sbuf[256 + row * 16 + n][head] = __expf(M1[i] - mrun[i]) * linv[i];
        }
        __syncthreads();
        bf16x16 P = postmix_a(sbuf, wpost);  // 16x32 post-mixed probs (A op)
#pragma unroll
        for (int dt = 0; dt < 4; ++dt) {
            const __bf16* vrow = vbase + (size_t)(dt * 16 + n) * SEQ + s0;
            bf16x16 vb = load_b_bf16(vrow, 0);
            ctx[dt] = WMMA_BF16(P, vb, ctx[dt]);
        }
        __syncthreads();
    }

    // ctx -> bf16 workspace, row-major [m = t*BSZ + b][e = head*64 + d]
#pragma unroll
    for (int dt = 0; dt < 4; ++dt)
#pragma unroll
        for (int i = 0; i < 8; ++i) {
            const int t = t0 + i + 8 * hi;
            const int col = head * HDIM + dt * 16 + n;
            ctx_ws[(size_t)(t * BSZ + b) * EMBED + col] = (__bf16)ctx[dt][i];
        }
}

// ---------------------------------------------------------------------------
extern "C" void kernel_launch(void* const* d_in, const int* in_sizes, int n_in,
                              void* d_out, int out_size, void* d_ws, size_t ws_size,
                              hipStream_t stream) {
    const float* query = (const float*)d_in[0];
    const float* key   = (const float*)d_in[1];
    const float* value = (const float*)d_in[2];
    const float* Wq = (const float*)d_in[3];
    const float* bq = (const float*)d_in[4];
    const float* Wk = (const float*)d_in[5];
    const float* bk = (const float*)d_in[6];
    const float* Wv = (const float*)d_in[7];
    const float* bv = (const float*)d_in[8];
    const float* Wpre  = (const float*)d_in[9];
    const float* Wpost = (const float*)d_in[10];
    const float* Wo = (const float*)d_in[11];
    const float* bo = (const float*)d_in[12];

    const size_t nqkv = (size_t)BSZ * HEADS * SEQ * HDIM;  // 4M bf16 each
    __bf16* q_ws   = (__bf16*)d_ws;
    __bf16* k_ws   = q_ws + nqkv;
    __bf16* vT_ws  = k_ws + nqkv;
    __bf16* ctx_ws = vT_ws + nqkv;                         // MROWS*EMBED bf16

    dim3 gemm_grid(EMBED / 256, MROWS / 128);  // (4, 32)
    proj_kernel<<<gemm_grid, 256, 0, stream>>>(query, Wq, bq, q_ws, 0.125f, 0);
    proj_kernel<<<gemm_grid, 256, 0, stream>>>(key,   Wk, bk, k_ws, 1.0f, 0);
    proj_kernel<<<gemm_grid, 256, 0, stream>>>(value, Wv, bv, vT_ws, 1.0f, 1);

    attn_kernel<<<BSZ * (SEQ / 16), 512, 0, stream>>>(q_ws, k_ws, vT_ws,
                                                      Wpre, Wpost, ctx_ws);

    out_proj_kernel<<<gemm_grid, 256, 0, stream>>>(ctx_ws, Wo, bo, (float*)d_out);
}
